// MultiHeadAttention_23398981828780
// MI455X (gfx1250) — matrix operand order repaired and checked by reading, not compiled, not run on hardware
//
#include <hip/hip_runtime.h>

typedef __attribute__((ext_vector_type(16))) __bf16 v16bf;
typedef __attribute__((ext_vector_type(8)))  float  v8f;

union FragU { v16bf v; uint4 q[2]; };

__device__ inline v16bf ldfrag(const __bf16* p0, const __bf16* p1) {
    FragU f;
    f.q[0] = *(const uint4*)p0;
    f.q[1] = *(const uint4*)p1;
    return f.v;
}

__device__ inline v8f vzero() {
    v8f z = {0.f, 0.f, 0.f, 0.f, 0.f, 0.f, 0.f, 0.f};
    return z;
}

__device__ inline __bf16 f2bf(float f) {
    unsigned u = __float_as_uint(f);
    u += 0x7FFFu + ((u >> 16) & 1u);          // round-to-nearest-even
    unsigned short s = (unsigned short)(u >> 16);
    return __builtin_bit_cast(__bf16, s);
}

__device__ inline unsigned pack2bf(float a, float b) {
    unsigned ua = __float_as_uint(a); ua += 0x7FFFu + ((ua >> 16) & 1u);
    unsigned ub = __float_as_uint(b); ub += 0x7FFFu + ((ub >> 16) & 1u);
    return (ua >> 16) | (ub & 0xFFFF0000u);
}

__device__ inline v8f wmma_bf16(v16bf a, v16bf b, v8f c) {
    return __builtin_amdgcn_wmma_f32_16x16x32_bf16(
        false, a, false, b, (short)0, c, false, false);
}

// ---- CDNA5 async memory->LDS copy (ASYNCcnt tracked) ----------------------
__device__ inline void async_b128(unsigned lds_addr, unsigned voff, const void* base) {
    asm volatile("global_load_async_to_lds_b128 %0, %1, %2"
                 :: "v"(lds_addr), "v"(voff), "s"(base) : "memory");
}
__device__ inline void wait_async() {
    asm volatile("s_wait_asynccnt 0x0" ::: "memory");
}
__device__ inline unsigned lds_off(const void* p) {
    return (unsigned)(size_t)p;   // LDS aperture keeps offset in low 32 bits
}

// ---- VALU 16-lane row reductions via v_permlane16_b32 ---------------------
__device__ inline float pl16(float v, unsigned lo, unsigned hi) {
    return __int_as_float(__builtin_amdgcn_permlane16(
        __float_as_int(v), __float_as_int(v), lo, hi, false, false));
}
__device__ inline float rowmax16(float v) {
    v = fmaxf(v, pl16(v, 0x67452301u, 0xEFCDAB89u));  // xor 1
    v = fmaxf(v, pl16(v, 0x54761032u, 0xDCFE98BAu));  // xor 2
    v = fmaxf(v, pl16(v, 0x32107654u, 0xBA98FEDCu));  // xor 4
    v = fmaxf(v, pl16(v, 0xFEDCBA98u, 0x76543210u));  // xor 8
    return v;
}
__device__ inline float rowsum16(float v) {
    v += pl16(v, 0x67452301u, 0xEFCDAB89u);
    v += pl16(v, 0x54761032u, 0xDCFE98BAu);
    v += pl16(v, 0x32107654u, 0xBA98FEDCu);
    v += pl16(v, 0xFEDCBA98u, 0x76543210u);
    return v;
}

// ---------------------------------------------------------------------------
// f32 -> bf16 elementwise convert (8 elems/thread, 16B stores)
// ---------------------------------------------------------------------------
__global__ void cvt_kernel(const float* __restrict__ in, __bf16* __restrict__ out, int n) {
    int i = (blockIdx.x * blockDim.x + threadIdx.x) * 8;
    const int stride = gridDim.x * blockDim.x * 8;
    for (; i < n; i += stride) {
        float4 a = *(const float4*)&in[i];
        float4 b = *(const float4*)&in[i + 4];
        uint4 o;
        o.x = pack2bf(a.x, a.y);
        o.y = pack2bf(a.z, a.w);
        o.z = pack2bf(b.x, b.y);
        o.w = pack2bf(b.z, b.w);
        *(uint4*)&out[i] = o;
    }
}

// ---------------------------------------------------------------------------
// f32 [K][N] -> bf16 [N][K] transpose (LDS tiled, 32x32)
// ---------------------------------------------------------------------------
__global__ __launch_bounds__(256) void transpose_cvt_kernel(
    const float* __restrict__ in, __bf16* __restrict__ out, int K, int N) {
    __shared__ float tile[32][33];
    const int n0 = blockIdx.x * 32, k0 = blockIdx.y * 32;
    const int x = threadIdx.x & 31, y = threadIdx.x >> 5;
#pragma unroll
    for (int r = 0; r < 4; r++)
        tile[y + 8 * r][x] = in[(size_t)(k0 + y + 8 * r) * N + n0 + x];
    __syncthreads();
#pragma unroll
    for (int r = 0; r < 4; r++)
        out[(size_t)(n0 + y + 8 * r) * K + k0 + x] = f2bf(tile[x][y + 8 * r]);
}

// ---------------------------------------------------------------------------
// bf16 GEMM: C[M,N] = A[M,K] * BT[N,K]^T + bias
// Block tile 64x256, 8 waves (2x4), wave tile 32x64, K-step 32.
// Async-to-LDS double-buffered pipeline, one barrier per K-step.
// MODE 0: qkv epilogue -> scatter bf16 into Q[B,H,T,D], K[B,H,T,D], Vt[B,H,D,T]
// MODE 1: proj epilogue -> f32 row-major output
// ---------------------------------------------------------------------------
template <int MODE>
__global__ __launch_bounds__(256) void gemm_bf16_kernel(
    const __bf16* __restrict__ A, const __bf16* __restrict__ BT,
    const float* __restrict__ bias,
    __bf16* __restrict__ Qo, __bf16* __restrict__ Ko, __bf16* __restrict__ Vto,
    float* __restrict__ Co,
    int M, int N, int Kd) {
    __shared__ __bf16 sA[2][64 * 32];    // row stride 32 bf16 = 64B (16B aligned)
    __shared__ __bf16 sB[2][256 * 32];

    const int tid = threadIdx.x;
    const int lane = tid & 31, wid = tid >> 5;
    const int half = lane >> 4, lm = lane & 15;
    const int wave_m = wid >> 2, wave_n = wid & 3;
    const int m0 = blockIdx.y * 64, n0 = blockIdx.x * 256;

    v8f acc[2][4];
#pragma unroll
    for (int i = 0; i < 2; i++)
#pragma unroll
        for (int j = 0; j < 4; j++) acc[i][j] = vzero();

    const int a_row = tid >> 2;
    const int a_ch = (tid & 3) * 8;

    auto issue = [&](int buf, int k) {
        // A tile: 64x32, one async b128 per thread
        async_b128(lds_off(&sA[buf][a_row * 32 + a_ch]),
                   (unsigned)(((size_t)(m0 + a_row) * Kd + k + a_ch) * 2), A);
        // B tile: 256x32, four async b128 per thread
        const unsigned ldb = lds_off(&sB[buf][tid * 32]);
        const unsigned vb = (unsigned)(((size_t)(n0 + tid) * Kd + k) * 2);
#pragma unroll
        for (int c = 0; c < 4; c++)
            async_b128(ldb + c * 16, vb + c * 16, BT);
    };

    issue(0, 0);
    const int nk = Kd >> 5;
    for (int kk = 0; kk < nk; kk++) {
        const int cur = kk & 1;
        wait_async();
        __syncthreads();
        if (kk + 1 < nk) issue(cur ^ 1, (kk + 1) << 5);

        v16bf af[2], bfg[4];
#pragma unroll
        for (int mi = 0; mi < 2; mi++) {
            // A layout: lane row = lm, 16 contiguous K starting at 16*half
            const __bf16* p = &sA[cur][(wave_m * 32 + mi * 16 + lm) * 32 + 16 * half];
            af[mi] = ldfrag(p, p + 8);
        }
#pragma unroll
        for (int ni = 0; ni < 4; ni++) {
            // B layout: lane col = lm, 16 contiguous K starting at 16*half
            const __bf16* p = &sB[cur][(wave_n * 64 + ni * 16 + lm) * 32 + 16 * half];
            bfg[ni] = ldfrag(p, p + 8);
        }
#pragma unroll
        for (int mi = 0; mi < 2; mi++)
#pragma unroll
            for (int ni = 0; ni < 4; ni++)
                acc[mi][ni] = wmma_bf16(af[mi], bfg[ni], acc[mi][ni]);
    }

    // Epilogue. C-layout: VGPR j -> row M = j + 8*half, col N = lm.
#pragma unroll
    for (int mi = 0; mi < 2; mi++) {
#pragma unroll
        for (int ni = 0; ni < 4; ni++) {
            const int n_col = n0 + wave_n * 64 + ni * 16 + lm;
            const float bv = bias[n_col];
            if constexpr (MODE == 0) {
                const int which = n_col >> 10;      // 0=Q 1=K 2=V
                const int c = n_col & 1023;
                const int h = c >> 6, d = c & 63;
#pragma unroll
                for (int j = 0; j < 8; j++) {
                    const int m_row = m0 + wave_m * 32 + mi * 16 + j + 8 * half;
                    const int bidx = m_row >> 11, t = m_row & 2047;
                    const int bh = bidx * 16 + h;
                    const float val = acc[mi][ni][j] + bv;
                    if (which == 2)
                        Vto[((size_t)bh * 64 + d) * 2048 + t] = f2bf(val);
                    else if (which == 1)
                        Ko[((size_t)bh * 2048 + t) * 64 + d] = f2bf(val);
                    else
                        Qo[((size_t)bh * 2048 + t) * 64 + d] = f2bf(val);
                }
            } else {
#pragma unroll
                for (int j = 0; j < 8; j++) {
                    const int m_row = m0 + wave_m * 32 + mi * 16 + j + 8 * half;
                    Co[(size_t)m_row * N + n_col] = acc[mi][ni][j] + bv;
                }
            }
        }
    }
}

// ---------------------------------------------------------------------------
// Causal flash attention. One block = 64 q rows of one (b,h); 4 waves x 16 rows.
// Q[B,H,T,D], K[B,H,T,D], Vt[B,H,D,T] bf16; Y[B,T,C] bf16.
// Async-to-LDS double-buffered K/V pipeline; VALU permlane row reductions.
// ---------------------------------------------------------------------------
__global__ __launch_bounds__(128) void attn_kernel(
    const __bf16* __restrict__ Q, const __bf16* __restrict__ K,
    const __bf16* __restrict__ Vt, __bf16* __restrict__ Y) {
    __shared__ __bf16 sK[2][32 * 72];     // 32 keys x 64 d (stride 144B, 16B-mult)
    __shared__ __bf16 sV[2][64 * 40];     // 64 d x 32 t   (stride 80B, 16B-mult)
    __shared__ __bf16 sP[4 * 16 * 40];    // per-wave 16x32 P tile

    const int tid = threadIdx.x, lane = tid & 31, wid = tid >> 5;
    const int half = lane >> 4, lm = lane & 15;
    const int qblk = blockIdx.x, bh = blockIdx.y;
    const int b = bh >> 4, h = bh & 15;
    const int q0 = qblk * 64, qw = q0 + wid * 16;

    const __bf16* Qb = Q + (size_t)bh * 2048 * 64;
    const __bf16* Kb = K + (size_t)bh * 2048 * 64;
    const __bf16* Vb = Vt + (size_t)bh * 64 * 2048;

    // Q A-fragments for d-steps 0,1 (register-resident)
    v16bf qf[2];
    {
        const __bf16* qr = Qb + (size_t)(qw + lm) * 64 + 16 * half;
        qf[0] = ldfrag(qr, qr + 8);
        qf[1] = ldfrag(qr + 32, qr + 40);
    }

    v8f O[4];
    float mrow[8], lpart[8];
#pragma unroll
    for (int i = 0; i < 4; i++) O[i] = vzero();
#pragma unroll
    for (int j = 0; j < 8; j++) { mrow[j] = -3.0e38f; lpart[j] = 0.f; }

    auto issue_kv = [&](int buf, int k0) {
#pragma unroll
        for (int r = 0; r < 2; r++) {
            const int idx = tid + r * 128;
            const int row = idx >> 3, ch = (idx & 7) * 8;
            async_b128(lds_off(&sK[buf][row * 72 + ch]),
                       (unsigned)(((size_t)(k0 + row) * 64 + ch) * 2), Kb);
        }
#pragma unroll
        for (int r = 0; r < 2; r++) {
            const int idx = tid + r * 128;
            const int row = idx >> 2, ch = (idx & 3) * 8;
            async_b128(lds_off(&sV[buf][row * 40 + ch]),
                       (unsigned)(((size_t)row * 2048 + k0 + ch) * 2), Vb);
        }
    };

    const float scale = 0.125f;     // 1/sqrt(64)
    const int nkb = qblk * 2 + 2;   // causal: keys up to q0+63

    issue_kv(0, 0);
    for (int kb = 0; kb < nkb; kb++) {
        const int cur = kb & 1;
        const int k0 = kb * 32;
        wait_async();
        __syncthreads();
        if (kb + 1 < nkb) issue_kv(cur ^ 1, (kb + 1) * 32);

        // ---- load ALL four K B-fragments up front (distinct regs, batched) ----
        const __bf16* pk00 = &sK[cur][lm * 72 + 16 * half];          // keys 0-15, d 0-31
        const __bf16* pk01 = &sK[cur][(16 + lm) * 72 + 16 * half];   // keys 16-31, d 0-31
        v16bf bk00 = ldfrag(pk00, pk00 + 8);
        v16bf bk01 = ldfrag(pk01, pk01 + 8);
        v16bf bk10 = ldfrag(pk00 + 32, pk00 + 40);                   // keys 0-15, d 32-63
        v16bf bk11 = ldfrag(pk01 + 32, pk01 + 40);                   // keys 16-31, d 32-63

        // S = Q * K^T : 16 q x 32 keys
        v8f s0 = vzero(), s1 = vzero();
        s0 = wmma_bf16(qf[0], bk00, s0);
        s1 = wmma_bf16(qf[0], bk01, s1);
        s0 = wmma_bf16(qf[1], bk10, s0);
        s1 = wmma_bf16(qf[1], bk11, s1);

        // online softmax: row = j + 8*half; its 16 cols live in one 16-lane half
#pragma unroll
        for (int j = 0; j < 8; j++) {
            const int rq = qw + j + 8 * half;
            float f0 = s0[j] * scale; if (k0 + lm > rq)      f0 = -1.0e9f;
            float f1 = s1[j] * scale; if (k0 + 16 + lm > rq) f1 = -1.0e9f;
            const float rmax = rowmax16(fmaxf(f0, f1));
            const float mnew = fmaxf(mrow[j], rmax);
            const float alpha = __expf(mrow[j] - mnew);
            const float p0 = __expf(f0 - mnew);
            const float p1 = __expf(f1 - mnew);
            lpart[j] = lpart[j] * alpha + p0 + p1;   // per-lane partial row sum
            mrow[j] = mnew;
#pragma unroll
            for (int ni = 0; ni < 4; ni++) O[ni][j] *= alpha;
            const int r = j + 8 * half;
            sP[(wid * 16 + r) * 40 + lm]      = f2bf(p0);
            sP[(wid * 16 + r) * 40 + 16 + lm] = f2bf(p1);
        }

        // P (16x32) as A-fragment via per-wave LDS bounce (same-wave LDS in order)
        const __bf16* pp = &sP[(wid * 16 + lm) * 40 + 16 * half];
        const v16bf pf = ldfrag(pp, pp + 8);

        // ---- load ALL four V B-fragments up front, then 4 WMMAs ----
        v16bf vf[4];
#pragma unroll
        for (int ni = 0; ni < 4; ni++) {
            const __bf16* vp = &sV[cur][(ni * 16 + lm) * 40 + 16 * half];
            vf[ni] = ldfrag(vp, vp + 8);
        }
#pragma unroll
        for (int ni = 0; ni < 4; ni++)
            O[ni] = wmma_bf16(pf, vf[ni], O[ni]);
    }

    // epilogue: reduce per-lane l partials once, normalize, write bf16 rows
#pragma unroll
    for (int j = 0; j < 8; j++) {
        const float l = rowsum16(lpart[j]);
        const float inv = 1.0f / l;
        const int t = qw + j + 8 * half;
        const size_t rowoff = ((size_t)b * 2048 + t) * 1024 + h * 64;
#pragma unroll
        for (int ni = 0; ni < 4; ni++)
            Y[rowoff + ni * 16 + lm] = f2bf(O[ni][j] * inv);
    }
}

// ---------------------------------------------------------------------------
extern "C" void kernel_launch(void* const* d_in, const int* in_sizes, int n_in,
                              void* d_out, int out_size, void* d_ws, size_t ws_size,
                              hipStream_t stream) {
    const float* x      = (const float*)d_in[0];
    const float* w_qkv  = (const float*)d_in[1];
    const float* b_qkv  = (const float*)d_in[2];
    const float* w_proj = (const float*)d_in[3];
    const float* b_proj = (const float*)d_in[4];
    float* out = (float*)d_out;

    // workspace carve-up (bf16 elements, all 16B aligned)
    const size_t NX = 4194304;          // B*T*C
    __bf16* ws     = (__bf16*)d_ws;
    __bf16* xb     = ws;                // 4194304
    __bf16* wqkvT  = xb + NX;           // 3145728
    __bf16* wprojT = wqkvT + 3145728;   // 1048576
    __bf16* Qw     = wprojT + 1048576;  // 4194304
    __bf16* Kw     = Qw + NX;           // 4194304
    __bf16* Vtw    = Kw + NX;           // 4194304
    __bf16* Y1     = Vtw + NX;          // 4194304  -> total ~50.3 MB

    cvt_kernel<<<1024, 256, 0, stream>>>(x, xb, (int)NX);
    transpose_cvt_kernel<<<dim3(3072 / 32, 1024 / 32), 256, 0, stream>>>(
        w_qkv, wqkvT, 1024, 3072);
    transpose_cvt_kernel<<<dim3(1024 / 32, 1024 / 32), 256, 0, stream>>>(
        w_proj, wprojT, 1024, 1024);

    // qkv = x @ w_qkv + b_qkv  (M=4096, N=3072, K=1024), scatter Q/K/Vt
    gemm_bf16_kernel<0><<<dim3(3072 / 256, 4096 / 64), 256, 0, stream>>>(
        xb, wqkvT, b_qkv, Qw, Kw, Vtw, nullptr, 4096, 3072, 1024);

    // causal flash attention -> Y1 [B,T,C] bf16
    attn_kernel<<<dim3(32, 32), 128, 0, stream>>>(Qw, Kw, Vtw, Y1);

    // out = Y1 @ w_proj + b_proj (M=4096, N=1024, K=1024), f32 output
    gemm_bf16_kernel<1><<<dim3(1024 / 256, 4096 / 64), 256, 0, stream>>>(
        Y1, wprojT, b_proj, nullptr, nullptr, nullptr, out, 4096, 1024, 1024);
}
